// Qwen3TTSTokenizerSingleCodebookVectorQuantization_12524124636033
// MI455X (gfx1250) — compile-verified, matrix operand order checked by reading, not run on hardware
//
#include <hip/hip_runtime.h>

typedef __attribute__((ext_vector_type(16))) __bf16 v16bf;
typedef __attribute__((ext_vector_type(8)))  __bf16 v8bf;
typedef __attribute__((ext_vector_type(8)))  float  v8f;

// f32 -> bf16 (RNE) -- let clang use the native gfx1250 convert
__device__ __forceinline__ __bf16 f2bf(float f) { return static_cast<__bf16>(f); }

// Build a 16x32 bf16 A/B fragment slice for this lane from an f32 row.
// lane<16 -> K {kb..kb+7, kb+16..kb+23}; lane>=16 -> K {kb+8..15, kb+24..31}
__device__ __forceinline__ v16bf frag_from_f32(const float* __restrict__ row, int kb, int kh) {
    v16bf r;
    const float* p0 = row + kb + kh;
    const float* p1 = row + kb + 16 + kh;
#pragma unroll
    for (int i = 0; i < 8; ++i) { r[i] = f2bf(p0[i]); r[8 + i] = f2bf(p1[i]); }
    return r;
}

__device__ __forceinline__ v16bf frag_from_bf16(const __bf16* __restrict__ row, int kb, int kh) {
    v8bf lo = *(const v8bf*)(row + kb + kh);
    v8bf hi = *(const v8bf*)(row + kb + 16 + kh);
    v16bf r;
#pragma unroll
    for (int i = 0; i < 8; ++i) { r[i] = lo[i]; r[8 + i] = hi[i]; }
    return r;
}

// ---------------------------------------------------------------------------
// Kernel 1: z = x @ W_in^T + b_in  -> bf16 z [16384, 256]
// ---------------------------------------------------------------------------
__global__ __launch_bounds__(256) void k_project_in(
    const float* __restrict__ x, const float* __restrict__ W_in,
    const float* __restrict__ b_in, __bf16* __restrict__ zbf)
{
    const int lane = threadIdx.x & 31;
    const int wave = blockIdx.x * 8 + (threadIdx.x >> 5);
    const int tt  = wave >> 4;          // token tile   0..1023
    const int ct  = wave & 15;          // channel tile 0..15
    const int l15 = lane & 15;
    const int kh  = (lane >> 4) << 3;   // 0 or 8

    const float* arow = x    + (size_t)(tt * 16 + l15) * 512;
    const float* brow = W_in + (size_t)(ct * 16 + l15) * 512;

    v8f acc = {};
#pragma unroll
    for (int kb = 0; kb < 512; kb += 32) {
        v16bf a = frag_from_f32(arow, kb, kh);
        v16bf b = frag_from_f32(brow, kb, kh);
        acc = __builtin_amdgcn_wmma_f32_16x16x32_bf16(false, a, false, b,
                                                      (short)0, acc, false, false);
    }
    const float bias = b_in[ct * 16 + l15];
    const int hi8 = (lane >> 4) * 8;
#pragma unroll
    for (int r = 0; r < 8; ++r) {
        int tok = tt * 16 + r + hi8;                 // C/D: M = r + 8*(lane>=16)
        zbf[(size_t)tok * 256 + ct * 16 + l15] = f2bf(acc[r] + bias);
    }
}

// ---------------------------------------------------------------------------
// Kernel 2: embed -> bf16, e_sq = ||embed_row||^2, W_out -> bf16.
// ---------------------------------------------------------------------------
__global__ __launch_bounds__(256) void k_prep(
    const float* __restrict__ embed, const float* __restrict__ W_out,
    __bf16* __restrict__ ebf, __bf16* __restrict__ wbf, float* __restrict__ e_sq)
{
    const int lane = threadIdx.x & 31;
    const int row  = blockIdx.x * 8 + (threadIdx.x >> 5);
    if (row < 8192) {
        const float* p = embed + (size_t)row * 256 + lane * 8;
        float s = 0.f;
        v8bf o;
#pragma unroll
        for (int i = 0; i < 8; ++i) { float v = p[i]; s += v * v; o[i] = f2bf(v); }
        *(v8bf*)(ebf + (size_t)row * 256 + lane * 8) = o;
#pragma unroll
        for (int off = 16; off > 0; off >>= 1) s += __shfl_xor(s, off, 32);
        if (lane == 0) e_sq[row] = s;
    } else if (row < 8192 + 512) {
        const int r2 = row - 8192;
        const float* p = W_out + (size_t)r2 * 256 + lane * 8;
        v8bf o;
#pragma unroll
        for (int i = 0; i < 8; ++i) o[i] = f2bf(p[i]);
        *(v8bf*)(wbf + (size_t)r2 * 256 + lane * 8) = o;
    }
}

// ---------------------------------------------------------------------------
// Kernel 3: argmax_c (2*z.e_c - ||e_c||^2) over 8192 codes.
// 8 waves/WG, each owns one 16-token tile (128 tokens per WG). The 16x256
// bf16 codebook tile (8 KB) is staged into LDS once per WG with
// global_load_async_to_lds_b128 (double buffered, ASYNCcnt + barriers) and
// shared by all 8 waves -> 8x less L2 traffic on the dominant GEMM.
// ---------------------------------------------------------------------------
#define SROW 272   // LDS row stride in bf16 elems (544 B: 16B-aligned, bank-spread)

__global__ __launch_bounds__(256) void k_argmax(
    const __bf16* __restrict__ zbf, const __bf16* __restrict__ ebf,
    const float* __restrict__ e_sq, int* __restrict__ ind)
{
    __shared__ __align__(16) __bf16 sb[2][16 * SROW];

    const int lane = threadIdx.x & 31;
    const int w    = threadIdx.x >> 5;                   // wave 0..7
    const int tt   = blockIdx.x * 8 + w;                 // token tile 0..1023
    const int l15  = lane & 15;
    const int kh   = (lane >> 4) << 3;

    // stage rows {2w, 2w+1} of code tile `ctile` into buffer `buf`
    auto issue = [&](int ctile, int buf) {
#pragma unroll
        for (int j = 0; j < 2; ++j) {
            const int r = 2 * w + j;
            const __bf16* g = ebf + (size_t)(ctile * 16 + r) * 256 + lane * 8;
            unsigned ldsa = (unsigned)(size_t)(&sb[buf][r * SROW + lane * 8]);
            asm volatile("global_load_async_to_lds_b128 %0, %1, off"
                         :: "v"(ldsa), "v"(g) : "memory");
        }
    };

    // preload A fragments (this wave's 16 tokens, full K=256)
    v16bf a[8];
    const __bf16* arow = zbf + (size_t)(tt * 16 + l15) * 256;
#pragma unroll
    for (int ks = 0; ks < 8; ++ks) a[ks] = frag_from_bf16(arow, ks * 32, kh);

    float bestv[8]; int besti[8];
#pragma unroll
    for (int r = 0; r < 8; ++r) { bestv[r] = -__builtin_inff(); besti[r] = 0; }

    issue(0, 0);
    issue(1, 1);

    for (int ctile = 0; ctile < 512; ++ctile) {
        // tiles <= ctile+1 issued (2 loads each, in-order completion):
        // asynccnt<=2 guarantees tile `ctile` has landed in LDS.
        asm volatile("s_wait_asynccnt 0x2" ::: "memory");
        __syncthreads();                                  // all 8 waves' rows visible

        const __bf16* brow = &sb[ctile & 1][l15 * SROW];
        v8f acc = {};
#pragma unroll
        for (int ks = 0; ks < 8; ++ks) {
            v8bf lo = *(const v8bf*)(brow + ks * 32 + kh);
            v8bf hi = *(const v8bf*)(brow + ks * 32 + 16 + kh);
            v16bf b;
#pragma unroll
            for (int i = 0; i < 8; ++i) { b[i] = lo[i]; b[8 + i] = hi[i]; }
            acc = __builtin_amdgcn_wmma_f32_16x16x32_bf16(false, a[ks], false, b,
                                                          (short)0, acc, false, false);
        }
        const int   code = ctile * 16 + l15;
        const float es   = e_sq[code];
#pragma unroll
        for (int r = 0; r < 8; ++r) {
            float s = 2.0f * acc[r] - es;
            if (s > bestv[r]) { bestv[r] = s; besti[r] = code; }  // strict > keeps first max
        }

        __syncthreads();                                  // everyone done with this buffer
        if (ctile + 2 < 512) issue(ctile + 2, ctile & 1); // overwrite it with tile+2
    }

    // reduce across the 16 lanes of each half-wave; ties -> smallest index
#pragma unroll
    for (int r = 0; r < 8; ++r) {
        float bv = bestv[r]; int bi = besti[r];
#pragma unroll
        for (int off = 8; off > 0; off >>= 1) {
            float ov = __shfl_xor(bv, off, 32);
            int   oi = __shfl_xor(bi, off, 32);
            if (ov > bv || (ov == bv && oi < bi)) { bv = ov; bi = oi; }
        }
        if (l15 == 0) {
            int tok = tt * 16 + r + ((lane >> 4) * 8);
            ind[tok] = bi;
        }
    }
}

// ---------------------------------------------------------------------------
// Kernel 4: out = embed[idx] @ W_out^T + b_out  (gathered bf16 A rows)
// ---------------------------------------------------------------------------
__global__ __launch_bounds__(256) void k_decode(
    const __bf16* __restrict__ ebf, const __bf16* __restrict__ wbf,
    const float* __restrict__ b_out, const int* __restrict__ ind,
    float* __restrict__ out)
{
    const int lane = threadIdx.x & 31;
    const int wave = blockIdx.x * 8 + (threadIdx.x >> 5);
    const int tt  = wave >> 5;          // token tile 0..1023
    const int dt  = wave & 31;          // dim tile   0..31
    const int l15 = lane & 15;
    const int kh  = (lane >> 4) << 3;

    const int code = ind[tt * 16 + l15];
    const __bf16* arow = ebf + (size_t)code * 256;
    const __bf16* brow = wbf + (size_t)(dt * 16 + l15) * 256;

    v8f acc = {};
#pragma unroll
    for (int ks = 0; ks < 8; ++ks) {
        v16bf a = frag_from_bf16(arow, ks * 32, kh);
        v16bf b = frag_from_bf16(brow, ks * 32, kh);
        acc = __builtin_amdgcn_wmma_f32_16x16x32_bf16(false, a, false, b,
                                                      (short)0, acc, false, false);
    }
    const float bias = b_out[dt * 16 + l15];
    const int hi8 = (lane >> 4) * 8;
#pragma unroll
    for (int r = 0; r < 8; ++r) {
        int tok = tt * 16 + r + hi8;
        out[(size_t)tok * 512 + dt * 16 + l15] = acc[r] + bias;
    }
}

// ---------------------------------------------------------------------------
extern "C" void kernel_launch(void* const* d_in, const int* in_sizes, int n_in,
                              void* d_out, int out_size, void* d_ws, size_t ws_size,
                              hipStream_t stream)
{
    (void)in_sizes; (void)n_in; (void)out_size; (void)ws_size;

    const float* x     = (const float*)d_in[0];   // [8,2048,512]
    const float* embed = (const float*)d_in[1];   // [8192,256]
    const float* W_in  = (const float*)d_in[2];   // [256,512]
    const float* b_in  = (const float*)d_in[3];   // [256]
    const float* W_out = (const float*)d_in[4];   // [512,256]
    const float* b_out = (const float*)d_in[5];   // [512]

    char* ws = (char*)d_ws;
    __bf16* zbf = (__bf16*)(ws);                                   // 8 MiB
    __bf16* ebf = (__bf16*)(ws + (size_t)8  * 1024 * 1024);        // 4 MiB
    __bf16* wbf = (__bf16*)(ws + (size_t)12 * 1024 * 1024);        // 256 KiB
    float*  esq = (float*) (ws + (size_t)12 * 1024 * 1024 + 256 * 1024); // 32 KiB

    int*   ind = (int*)d_out;                 // [8,2048] int32
    float* out = (float*)d_out + 16384;       // [8,2048,512] f32

    k_project_in<<<2048, 256, 0, stream>>>(x, W_in, b_in, zbf);
    k_prep      <<<1088, 256, 0, stream>>>(embed, W_out, ebf, wbf, esq);
    k_argmax    <<<128,  256, 0, stream>>>(zbf, ebf, esq, ind);
    k_decode    <<<4096, 256, 0, stream>>>(ebf, wbf, b_out, ind, out);
}
